// BioGNN_28467043238643
// MI455X (gfx1250) — matrix operand order repaired and checked by reading, not compiled, not run on hardware
//
#include <hip/hip_runtime.h>
#include <math.h>

#define NN 20000
#define EE 320000
#define ETOT 340000   // EE + NN self-loops (virtual, not materialized)
#define GG 50

typedef __attribute__((ext_vector_type(16))) __bf16 v16bf;
typedef __attribute__((ext_vector_type(8)))  __bf16 v8bf;
typedef __attribute__((ext_vector_type(8)))  float  v8f;

// ---------------- helpers ----------------

__device__ __forceinline__ unsigned short f2bf(float f) {
    unsigned int u = __float_as_uint(f);
    u = (u + 0x7FFFu + ((u >> 16) & 1u)) >> 16;   // round-to-nearest-even
    return (unsigned short)u;
}

__device__ __forceinline__ void edge_sd(const int* __restrict__ ei, int e, int& s, int& d) {
    if (e < EE) { s = ei[e]; d = ei[EE + e]; }
    else        { s = e - EE; d = e - EE; }       // self loops appended
}

__device__ __forceinline__ void atomicMaxF(float* addr, float val) {
    if (val >= 0.0f) atomicMax((int*)addr, __float_as_int(val));
    else             atomicMin((unsigned int*)addr, __float_as_uint(val));
}

// ---------------- generic fill ----------------

__global__ void k_fill(float* __restrict__ p, float v, long long n) {
    long long i = (long long)blockIdx.x * blockDim.x + threadIdx.x;
    if (i < n) p[i] = v;
}

// ---------------- layer 1 projection: hp[n,k] = x[n] * W1[k] ----------------

__global__ void k_hp1(const float* __restrict__ x, const float* __restrict__ W1,
                      float* __restrict__ hp) {
    long long i = (long long)blockIdx.x * blockDim.x + threadIdx.x;
    if (i >= (long long)NN * 512) return;
    hp[i] = x[i >> 9] * W1[i & 511];
}

// ---------------- fp32 -> bf16 weight convert ----------------

__global__ void k_f2bf(const float* __restrict__ src, unsigned short* __restrict__ dst, int n) {
    int i = blockIdx.x * blockDim.x + threadIdx.x;
    if (i < n) dst[i] = f2bf(src[i]);
}

// ---------------- WMMA bf16 GEMM: C[M,O] = A[M,K] * B[O,K]^T ----------------
// One wave per 16x16 output tile; K stepped by 32 (v_wmma_f32_16x16x32_bf16).
// Fragment layouts per CDNA5 ISA 7.12.2 (wave32).

__global__ __launch_bounds__(256)
void k_gemm_bf16(const unsigned short* __restrict__ A16,
                 const unsigned short* __restrict__ B16,
                 float* __restrict__ C, int M, int K, int O) {
    int wave = threadIdx.x >> 5;
    int lane = threadIdx.x & 31;
    int gw = blockIdx.x * 8 + wave;
    int mtiles = M >> 4, otiles = O >> 4;
    if (gw >= mtiles * otiles) return;          // wave-uniform: EXEC stays all-ones
    int mt = gw % mtiles, ot = gw / mtiles;
    int m0 = mt << 4,    o0 = ot << 4;
    int half = lane >> 4, l = lane & 15;

    const __bf16* A = (const __bf16*)A16;
    const __bf16* B = (const __bf16*)B16;
    // A frag: row M = m0 + l; lane half 0 -> K {kb+0..7, kb+16..23}, half 1 -> {kb+8..15, kb+24..31}
    const __bf16* Abase = A + (size_t)(m0 + l) * K;
    // B frag: col N = o0 + l; half 0 -> K {kb+0..7, kb+16..23}, half 1 -> {kb+8..15, kb+24..31}
    const __bf16* Bbase = B + (size_t)(o0 + l) * K;

    v8f acc = {};
    for (int kb = 0; kb < K; kb += 32) {
        if (kb + 32 < K) {
            __builtin_prefetch(Abase + kb + 32 + half * 8, 0, 1);
            __builtin_prefetch(Bbase + kb + 32, 0, 1);
        }
        v8bf x0 = *(const v8bf*)(Abase + kb + half * 8);
        v8bf x1 = *(const v8bf*)(Abase + kb + 16 + half * 8);
        v16bf a;
#pragma unroll
        for (int i = 0; i < 8; i++) { a[i] = x0[i]; a[i + 8] = x1[i]; }
        v8bf y0 = *(const v8bf*)(Bbase + kb + half * 8);
        v8bf y1 = *(const v8bf*)(Bbase + kb + 16 + half * 8);
        v16bf b;
#pragma unroll
        for (int i = 0; i < 8; i++) { b[i] = y0[i]; b[i + 8] = y1[i]; }
        acc = __builtin_amdgcn_wmma_f32_16x16x32_bf16(
            /*neg_a=*/false, a, /*neg_b=*/false, b,
            /*c_mod=*/(short)0, acc, /*reuse_a=*/false, /*reuse_b=*/false);
    }
    // C/D layout: VGPR r -> row m0 + r + 8*half, col o0 + l
    float* Crow = C + (size_t)(m0 + half * 8) * O + o0 + l;
#pragma unroll
    for (int r = 0; r < 8; r++) Crow[(size_t)r * O] = acc[r];
}

// ---------------- attention coefficients a_s[n,h], a_d[n,h] ----------------

__global__ void k_att(const float* __restrict__ hp, const float* __restrict__ asw,
                      const float* __restrict__ adw, float* __restrict__ a_s,
                      float* __restrict__ a_d, int H, int C) {
    int t = blockIdx.x * blockDim.x + threadIdx.x;
    if (t >= NN * H) return;
    int n = t / H, h = t - n * H;
    const float* row = hp + (size_t)n * H * C + h * C;
    const float* aw = asw + h * C;
    const float* dw = adw + h * C;
    float s = 0.f, d = 0.f;
    for (int c = 0; c < C; c++) { float v = row[c]; s += v * aw[c]; d += v * dw[c]; }
    a_s[t] = s; a_d[t] = d;
}

// ---------------- edge passes: segment max / exp-sum / normalize / aggregate ----------------

__device__ __forceinline__ float leaky(float x) { return x > 0.f ? x : 0.2f * x; }

__global__ void k_edge_max(const int* __restrict__ ei, const float* __restrict__ a_s,
                           const float* __restrict__ a_d, float* __restrict__ mbuf, int H) {
    int t = blockIdx.x * blockDim.x + threadIdx.x;
    if (t >= ETOT * H) return;
    int e = t / H, h = t - e * H;
    int s, d; edge_sd(ei, e, s, d);
    float lg = leaky(a_s[s * H + h] + a_d[d * H + h]);
    atomicMaxF(mbuf + d * H + h, lg);
}

__global__ void k_edge_ex(const int* __restrict__ ei, const float* __restrict__ a_s,
                          const float* __restrict__ a_d, const float* __restrict__ mbuf,
                          float* __restrict__ den, float* __restrict__ alpha, int H) {
    int t = blockIdx.x * blockDim.x + threadIdx.x;
    if (t >= ETOT * H) return;
    int e = t / H, h = t - e * H;
    int s, d; edge_sd(ei, e, s, d);
    float lg = leaky(a_s[s * H + h] + a_d[d * H + h]);
    float ex = expf(lg - mbuf[d * H + h]);
    alpha[t] = ex;
    atomicAdd(den + d * H + h, ex);
}

__global__ void k_edge_norm(const int* __restrict__ ei, const float* __restrict__ den,
                            float* __restrict__ alpha, int H) {
    int t = blockIdx.x * blockDim.x + threadIdx.x;
    if (t >= ETOT * H) return;
    int e = t / H, h = t - e * H;
    int s, d; edge_sd(ei, e, s, d);
    alpha[t] = alpha[t] / den[d * H + h];
}

__global__ void k_edge_aggr(const int* __restrict__ ei, const float* __restrict__ alpha,
                            const float* __restrict__ hp, float* __restrict__ out,
                            int H, int C) {
    int F = H * C;
    long long t = (long long)blockIdx.x * blockDim.x + threadIdx.x;
    if (t >= (long long)ETOT * F) return;
    int e = (int)(t / F);
    int j = (int)(t - (long long)e * F);
    int h = j / C;
    int s, d; edge_sd(ei, e, s, d);
    float w = alpha[(size_t)e * H + h];
    atomicAdd(out + (size_t)d * F + j, hp[(size_t)s * F + j] * w);
}

// ---------------- BatchNorm (F=512) ----------------

__global__ __launch_bounds__(256)
void k_bn_stats(const float* __restrict__ h, float* __restrict__ sum,
                float* __restrict__ sqs) {
    int r0 = blockIdx.x * 40;
    int c = threadIdx.x;                      // channels c and c+256
    float s0 = 0.f, q0 = 0.f, s1 = 0.f, q1 = 0.f;
    int rend = min(r0 + 40, NN);
    for (int r = r0; r < rend; r++) {
        const float* row = h + (size_t)r * 512;
        float v0 = row[c], v1 = row[c + 256];
        s0 += v0; q0 += v0 * v0; s1 += v1; q1 += v1 * v1;
    }
    atomicAdd(sum + c, s0);       atomicAdd(sqs + c, q0);
    atomicAdd(sum + c + 256, s1); atomicAdd(sqs + c + 256, q1);
}

__global__ void k_bn_final(const float* __restrict__ sum, const float* __restrict__ sqs,
                           const float* __restrict__ g, float* __restrict__ mu,
                           float* __restrict__ scale) {
    int c = blockIdx.x * blockDim.x + threadIdx.x;
    if (c >= 512) return;
    float m = sum[c] * (1.0f / NN);
    float v = sqs[c] * (1.0f / NN) - m * m;
    mu[c] = m;
    scale[c] = g[c] * rsqrtf(v + 1e-5f);
}

__global__ void k_bn_apply(const float* __restrict__ h, const float* __restrict__ mu,
                           const float* __restrict__ scale, const float* __restrict__ beta,
                           unsigned short* __restrict__ out16) {
    long long i = (long long)blockIdx.x * blockDim.x + threadIdx.x;
    if (i >= (long long)NN * 512) return;
    int c = (int)(i & 511);
    float v = (h[i] - mu[c]) * scale[c] + beta[c];
    out16[i] = f2bf(fmaxf(v, 0.0f));          // ReLU fused, bf16 for next WMMA GEMM
}

// ---------------- layer-3 tail: bias+relu, pooling, MLP head ----------------

__global__ void k_bias_relu(float* __restrict__ h, const float* __restrict__ b) {
    long long i = (long long)blockIdx.x * blockDim.x + threadIdx.x;
    if (i >= (long long)NN * 128) return;
    h[i] = fmaxf(h[i] + b[i & 127], 0.0f);
}

__global__ void k_pool(const float* __restrict__ h, const int* __restrict__ batch,
                       float* __restrict__ pool, float* __restrict__ cnt) {
    long long i = (long long)blockIdx.x * blockDim.x + threadIdx.x;
    if (i >= (long long)NN * 128) return;
    int n = (int)(i >> 7), c = (int)(i & 127);
    int g = batch[n];
    atomicAdd(pool + g * 128 + c, h[i]);
    if (c == 0) atomicAdd(cnt + g, 1.0f);
}

__global__ __launch_bounds__(64)
void k_head(const float* __restrict__ pool, const float* __restrict__ cnt,
            const float* __restrict__ lw1, const float* __restrict__ lb1,
            const float* __restrict__ lw2, const float* __restrict__ lb2,
            float* __restrict__ out) {
    __shared__ float t1[64];
    int g = blockIdx.x, j = threadIdx.x;
    float inv = 1.0f / fmaxf(cnt[g], 1.0f);
    float acc = lb1[j];
    for (int c = 0; c < 128; c++) acc += pool[g * 128 + c] * inv * lw1[j * 128 + c];
    t1[j] = fmaxf(acc, 0.0f);
    __syncthreads();
    if (j < 5) {
        float o = lb2[j];
        for (int c = 0; c < 64; c++) o += t1[c] * lw2[j * 64 + c];
        out[g * 5 + j] = o;
    }
}

// ---------------- host orchestration ----------------

extern "C" void kernel_launch(void* const* d_in, const int* in_sizes, int n_in,
                              void* d_out, int out_size, void* d_ws, size_t ws_size,
                              hipStream_t stream) {
    (void)in_sizes; (void)n_in; (void)out_size; (void)ws_size;

    const float* x    = (const float*)d_in[0];
    const int*   ei   = (const int*)d_in[1];
    const int*   batch= (const int*)d_in[2];
    const float* W1   = (const float*)d_in[3];
    const float* as1  = (const float*)d_in[4];
    const float* ad1  = (const float*)d_in[5];
    const float* g1   = (const float*)d_in[7];
    const float* be1  = (const float*)d_in[8];
    const float* W2   = (const float*)d_in[9];
    const float* as2  = (const float*)d_in[10];
    const float* ad2  = (const float*)d_in[11];
    const float* g2   = (const float*)d_in[13];
    const float* be2  = (const float*)d_in[14];
    const float* W3   = (const float*)d_in[15];
    const float* as3  = (const float*)d_in[16];
    const float* ad3  = (const float*)d_in[17];
    const float* b3   = (const float*)d_in[18];
    const float* lw1  = (const float*)d_in[19];
    const float* lb1  = (const float*)d_in[20];
    const float* lw2  = (const float*)d_in[21];
    const float* lb2  = (const float*)d_in[22];
    // b1/b2 are mathematically absorbed by BatchNorm mean-subtraction; omitted.

    char* base = (char*)d_ws;
    size_t off = 0;
    auto alloc = [&](size_t bytes) -> void* {
        void* p = base + off;
        off = (off + bytes + 255) & ~(size_t)255;
        return p;
    };
    float*          hp      = (float*)alloc((size_t)NN * 512 * 4);
    float*          agg     = (float*)alloc((size_t)NN * 512 * 4);
    unsigned short* h16     = (unsigned short*)alloc((size_t)NN * 512 * 2);
    unsigned short* w16     = (unsigned short*)alloc((size_t)512 * 512 * 2);
    float*          a_s     = (float*)alloc((size_t)NN * 4 * 4);
    float*          a_d     = (float*)alloc((size_t)NN * 4 * 4);
    float*          mbuf    = (float*)alloc((size_t)NN * 4 * 4);
    float*          den     = (float*)alloc((size_t)NN * 4 * 4);
    float*          alpha   = (float*)alloc((size_t)ETOT * 4 * 4);
    float*          bnsum   = (float*)alloc(1024 * 4);         // [0:512)=sum, [512:1024)=sumsq
    float*          bnmu    = (float*)alloc(512 * 4);
    float*          bnscale = (float*)alloc(512 * 4);
    float*          pool    = (float*)alloc((GG * 128 + GG) * 4); // sums then counts

    auto g1d = [](long long n) { return dim3((unsigned)((n + 255) / 256)); };

    auto run_edges = [&](const float* hpL, float* aggL, int H, int C,
                         const float* asw, const float* adw) {
        int F = H * C;
        k_att      <<<g1d((long long)NN * H),   256, 0, stream>>>(hpL, asw, adw, a_s, a_d, H, C);
        k_fill     <<<g1d((long long)NN * H),   256, 0, stream>>>(mbuf, -INFINITY, (long long)NN * H);
        k_fill     <<<g1d((long long)NN * H),   256, 0, stream>>>(den, 0.0f, (long long)NN * H);
        k_edge_max <<<g1d((long long)ETOT * H), 256, 0, stream>>>(ei, a_s, a_d, mbuf, H);
        k_edge_ex  <<<g1d((long long)ETOT * H), 256, 0, stream>>>(ei, a_s, a_d, mbuf, den, alpha, H);
        k_edge_norm<<<g1d((long long)ETOT * H), 256, 0, stream>>>(ei, den, alpha, H);
        k_fill     <<<g1d((long long)NN * F),   256, 0, stream>>>(aggL, 0.0f, (long long)NN * F);
        k_edge_aggr<<<g1d((long long)ETOT * F), 256, 0, stream>>>(ei, alpha, hpL, aggL, H, C);
    };

    auto run_bn = [&](const float* aggL, const float* g, const float* be, unsigned short* out16) {
        k_fill    <<<g1d(1024), 256, 0, stream>>>(bnsum, 0.0f, 1024);
        k_bn_stats<<<dim3((NN + 39) / 40), 256, 0, stream>>>(aggL, bnsum, bnsum + 512);
        k_bn_final<<<dim3(2), 256, 0, stream>>>(bnsum, bnsum + 512, g, bnmu, bnscale);
        k_bn_apply<<<g1d((long long)NN * 512), 256, 0, stream>>>(aggL, bnmu, bnscale, be, out16);
    };

    auto run_gemm = [&](const unsigned short* A16, const unsigned short* B16, float* Cout, int O) {
        int tiles = (NN / 16) * (O / 16);
        k_gemm_bf16<<<dim3((tiles + 7) / 8), 256, 0, stream>>>(A16, B16, Cout, NN, 512, O);
    };

    // ---- Layer 1: outer-product projection (in=1), GAT, BN+ReLU -> bf16 ----
    k_hp1<<<g1d((long long)NN * 512), 256, 0, stream>>>(x, W1, hp);
    run_edges(hp, agg, 4, 128, as1, ad1);
    run_bn(agg, g1, be1, h16);

    // ---- Layer 2: WMMA GEMM (512x512), GAT, BN+ReLU -> bf16 ----
    k_f2bf<<<g1d(512 * 512), 256, 0, stream>>>(W2, w16, 512 * 512);
    run_gemm(h16, w16, hp, 512);
    run_edges(hp, agg, 4, 128, as2, ad2);
    run_bn(agg, g2, be2, h16);

    // ---- Layer 3: WMMA GEMM (128x512), GAT (H=1), bias+ReLU ----
    k_f2bf<<<g1d(128 * 512), 256, 0, stream>>>(W3, w16, 128 * 512);
    run_gemm(h16, w16, hp, 128);
    run_edges(hp, agg, 1, 128, as3, ad3);
    k_bias_relu<<<g1d((long long)NN * 128), 256, 0, stream>>>(agg, b3);

    // ---- Global mean pool + MLP head ----
    k_fill<<<g1d(GG * 128 + GG), 256, 0, stream>>>(pool, 0.0f, GG * 128 + GG);
    k_pool<<<g1d((long long)NN * 128), 256, 0, stream>>>(agg, batch, pool, pool + GG * 128);
    k_head<<<dim3(GG), 64, 0, stream>>>(pool, pool + GG * 128, lw1, lb1, lw2, lb2, (float*)d_out);
}
